// Attention_55619826483814
// MI455X (gfx1250) — matrix operand order repaired and checked by reading, not compiled, not run on hardware
//
#include <hip/hip_runtime.h>
#include <hip/hip_bf16.h>

typedef _Float16 h16;
typedef __attribute__((ext_vector_type(8)))  _Float16 v8h;
typedef __attribute__((ext_vector_type(16))) _Float16 v16h;
typedef __attribute__((ext_vector_type(8)))  float    v8f;
typedef __attribute__((ext_vector_type(4)))  int      v4i;

#define B_       4
#define C_       256
#define N_       4096
#define HEADS_   4
#define DH_      32
#define HID_     128
#define QKV_R    384
#define SCALE_   0.17677669529663687f   // 32^-0.5
#define EPS_     1e-5f

#define WMMA_F16(a, b, c) \
  __builtin_amdgcn_wmma_f32_16x16x32_f16(false, (a), false, (b), (short)0, (c), false, false)

// gfx1250 async global->LDS copy path (ASYNCcnt-tracked), guarded per toolchain.
#if defined(__has_builtin)
#if __has_builtin(__builtin_amdgcn_global_load_async_to_lds_b128) && \
    __has_builtin(__builtin_amdgcn_s_wait_asynccnt)
#define HAVE_ASYNC_LDS 1
#endif
#endif
#ifndef HAVE_ASYNC_LDS
#define HAVE_ASYNC_LDS 0
#endif

static __device__ inline v16h make16(v8h lo, v8h hi) {
  return __builtin_shufflevector(lo, hi, 0,1,2,3,4,5,6,7,8,9,10,11,12,13,14,15);
}

// A-matrix lane load (16x32 f16). rowbase = &row[lane&15] at current k-base.
// lanes 0-15 hold K {0..15}; lanes 16-31 hold {16..31}.
static __device__ inline v16h load_a(const h16* rowbase, int lane) {
  const int koff = (lane & 16) ? 16 : 0;
  v8h lo = *(const v8h*)(rowbase + koff);
  v8h hi = *(const v8h*)(rowbase + koff + 8);
  return make16(lo, hi);
}

// B-matrix lane load (32x16 f16): p = this lane's 16 contiguous K values.
static __device__ inline v16h load_b(const h16* p) {
  v8h lo = *(const v8h*)(p);
  v8h hi = *(const v8h*)(p + 8);
  return make16(lo, hi);
}

// stage 16 bytes global -> LDS (async when available)
static __device__ inline void stage16(const h16* gsrc, h16* ldst) {
#if HAVE_ASYNC_LDS
  __builtin_amdgcn_global_load_async_to_lds_b128(
      (__attribute__((address_space(1))) v4i*)gsrc,
      (__attribute__((address_space(3))) v4i*)ldst, 0, 0);
#else
  *(v8h*)ldst = *(const v8h*)gsrc;
#endif
}

static __device__ inline void stage_wait_prev(bool more_in_flight) {
#if HAVE_ASYNC_LDS
  if (more_in_flight) __builtin_amdgcn_s_wait_asynccnt(2);
  else                __builtin_amdgcn_s_wait_asynccnt(0);
#else
  (void)more_in_flight;
#endif
}

// ---------------- prep: zero stats, convert weights to f16 ----------------
__global__ __launch_bounds__(256) void prep_kernel(const float* __restrict__ wq,
                                                   const float* __restrict__ wo,
                                                   h16* __restrict__ wq16,
                                                   h16* __restrict__ wo16,
                                                   float* __restrict__ stats) {
  const int i = blockIdx.x * blockDim.x + threadIdx.x;
  const int stride = gridDim.x * blockDim.x;
  if (i < 2 * B_) stats[i] = 0.0f;
  for (int j = i; j < QKV_R * C_; j += stride) wq16[j] = (h16)wq[j];
  for (int j = i; j < C_ * HID_; j += stride) wo16[j] = (h16)wo[j];
}

// ---------------- per-batch sum / sumsq over 1M elements ----------------
__global__ __launch_bounds__(256) void stats_kernel(const float* __restrict__ x,
                                                    float* __restrict__ stats) {
  __shared__ float ss[256], sq[256];
  const int b = blockIdx.y;
  const float* xb = x + (size_t)b * C_ * N_;
  const int tid = threadIdx.x;
  const int base = blockIdx.x * 16384 + tid;
  float s = 0.f, q = 0.f;
  #pragma unroll 4
  for (int i = 0; i < 64; ++i) {
    float v = xb[base + i * 256];
    s += v; q += v * v;
  }
  ss[tid] = s; sq[tid] = q;
  __syncthreads();
  for (int off = 128; off > 0; off >>= 1) {
    if (tid < off) { ss[tid] += ss[tid + off]; sq[tid] += sq[tid + off]; }
    __syncthreads();
  }
  if (tid == 0) {
    atomicAdd(&stats[b * 2 + 0], ss[0]);
    atomicAdd(&stats[b * 2 + 1], sq[0]);
  }
}

// ---------------- LayerNorm -> xnT f16 pixel-major [b][n][256] ----------------
__global__ __launch_bounds__(256) void normalize_kernel(const float* __restrict__ x,
                                                        const float* __restrict__ gamma,
                                                        const float* __restrict__ beta,
                                                        const float* __restrict__ stats,
                                                        h16* __restrict__ xnT) {
  const int idx = blockIdx.x * 256 + threadIdx.x;  // b*N + n
  const int b = idx >> 12;
  const int n = idx & (N_ - 1);
  const float inv = 1.0f / (float)(C_ * N_);
  const float mu = stats[b * 2 + 0] * inv;
  const float var = stats[b * 2 + 1] * inv - mu * mu;
  const float rstd = rsqrtf(var + EPS_);
  const float* xb = x + (size_t)b * C_ * N_ + n;
  h16* op = xnT + (size_t)idx * C_;
  for (int c0 = 0; c0 < C_; c0 += 8) {
    v8h o;
    #pragma unroll
    for (int j = 0; j < 8; ++j) {
      const int c = c0 + j;
      const float v = xb[(size_t)c * N_];
      o[j] = (h16)((v - mu) * rstd * gamma[c] + beta[c]);
    }
    *(v8h*)(op + c0) = o;
  }
}

// ---------------- QKV projection: [384,256] x [256,4096] per batch ----------------
// Writes Q (scaled) and K row-major [bh][n][32]; V channel-major [bh][32][n].
__global__ __launch_bounds__(128) void qkv_gemm_kernel(const h16* __restrict__ wq16,
                                                       const h16* __restrict__ xnT,
                                                       h16* __restrict__ Qb,
                                                       h16* __restrict__ Kb,
                                                       h16* __restrict__ Vb) {
  const int lane = threadIdx.x & 31;
  const int wid  = threadIdx.x >> 5;
  const int tile = blockIdx.x * 4 + wid;          // 4 * 24 * 256 = 24576 tiles
  const int b   = tile / (24 * 256);
  const int rem = tile - b * (24 * 256);
  const int ot  = rem >> 8;                        // 0..23
  const int nt  = rem & 255;                       // 0..255
  const int r15 = lane & 15;
  const int hi16 = lane >> 4;

  const h16* X = xnT + (size_t)b * N_ * C_;
  v8f acc = {};
  #pragma unroll
  for (int ks = 0; ks < 8; ++ks) {
    const int kb = ks * 32;
    const h16* arow = wq16 + (size_t)(ot * 16 + r15) * C_ + kb;
    v16h a = load_a(arow, lane);
    const h16* bp = X + (size_t)(nt * 16 + r15) * C_ + kb + hi16 * 16;
    v16h bm = load_b(bp);
    acc = WMMA_F16(a, bm, acc);
  }
  #pragma unroll
  for (int r = 0; r < 8; ++r) {
    const int o = ot * 16 + r + hi16 * 8;          // output channel 0..383
    const int n = nt * 16 + r15;
    const int sect = o >> 7;                       // 0=q 1=k 2=v
    const int within = o & 127;
    const int h = within >> 5;
    const int c = within & 31;
    const size_t bh = (size_t)(b * HEADS_ + h);
    const float v = acc[r];
    if (sect == 0)      Qb[(bh * N_ + n) * DH_ + c] = (h16)(v * SCALE_);
    else if (sect == 1) Kb[(bh * N_ + n) * DH_ + c] = (h16)v;
    else                Vb[(bh * DH_ + c) * N_ + n] = (h16)v;
  }
}

// ---------------- flash attention ----------------
// Block = 4 waves, one (b,h), 4 consecutive 16-query tiles. K/V tiles for each
// 32-key step are staged once per block into double-buffered LDS via async
// global->LDS copies; all 4 waves feed WMMA from LDS (ds_load_b128).
__global__ __launch_bounds__(128) void flash_attn_kernel(const h16* __restrict__ Qb,
                                                         const h16* __restrict__ Kb,
                                                         const h16* __restrict__ Vb,
                                                         h16* __restrict__ Ob) {
  __shared__ h16 kbuf[2][32 * 32];   // [key][ch]
  __shared__ h16 vbuf[2][32 * 32];   // [ch][key]
  __shared__ h16 pbuf[4][16 * 32];   // per-wave P tile

  const int tid  = threadIdx.x;
  const int lane = tid & 31;
  const int wid  = tid >> 5;
  const int bh   = blockIdx.x >> 6;                 // 0..15
  const int qt   = (blockIdx.x & 63) * 4 + wid;     // 0..255
  const int r15  = lane & 15;
  const int hi16 = lane >> 4;

  const h16* Qh = Qb + (size_t)bh * N_ * DH_;
  const h16* Kh = Kb + (size_t)bh * N_ * DH_;
  const h16* Vh = Vb + (size_t)bh * DH_ * N_;
  h16* plds = pbuf[wid];

  // staging indices: 128 threads x 16B cover one 32x32 f16 tile
  const int srow = tid >> 2;               // 0..31
  const int soff = (tid & 3) * 8;          // 0,8,16,24

  const h16* qrow = Qh + (size_t)(qt * 16 + r15) * DH_;
  const v16h qa = load_a(qrow, lane);      // 16x32 A tile, scale pre-applied

  float m[8], l[8];
  v8f o0 = {}, o1 = {};
  #pragma unroll
  for (int r = 0; r < 8; ++r) { m[r] = -3.0e38f; l[r] = 0.0f; }

  const int nk = N_ / 32;
  // prologue: stage key-step 0 into buffer 0
  stage16(Kh + (size_t)srow * DH_ + soff, kbuf[0] + srow * 32 + soff);
  stage16(Vh + (size_t)srow * N_ + soff, vbuf[0] + srow * 32 + soff);

  #pragma unroll 1
  for (int kt = 0; kt < nk; ++kt) {
    const int cur = kt & 1;
    const bool more = (kt + 1) < nk;
    if (more) {
      const int kb2 = (kt + 1) * 32;
      stage16(Kh + (size_t)(kb2 + srow) * DH_ + soff, kbuf[cur ^ 1] + srow * 32 + soff);
      stage16(Vh + (size_t)srow * N_ + kb2 + soff,    vbuf[cur ^ 1] + srow * 32 + soff);
    }
    stage_wait_prev(more);   // previous stage's 2 per-thread async copies landed
    __syncthreads();         // visible to all waves

    const h16* kl = kbuf[cur];
    const h16* vl = vbuf[cur];
    // S = Q x K^T : two 16x16 key tiles from LDS
    v16h b0 = load_b(kl + (size_t)r15 * 32 + hi16 * 16);
    v16h b1 = load_b(kl + (size_t)(16 + r15) * 32 + hi16 * 16);
    v8f s0 = {}, s1 = {};
    s0 = WMMA_F16(qa, b0, s0);
    s1 = WMMA_F16(qa, b1, s1);

    // online softmax per row (rows live on half-wave lane groups of 16)
    #pragma unroll
    for (int r = 0; r < 8; ++r) {
      float a0 = s0[r], a1 = s1[r];
      float mx = fmaxf(a0, a1);
      #pragma unroll
      for (int off = 8; off > 0; off >>= 1) mx = fmaxf(mx, __shfl_xor(mx, off, 16));
      const float nm = fmaxf(m[r], mx);
      const float corr = __expf(m[r] - nm);
      const float p0 = __expf(a0 - nm);
      const float p1 = __expf(a1 - nm);
      float rs = p0 + p1;
      #pragma unroll
      for (int off = 8; off > 0; off >>= 1) rs += __shfl_xor(rs, off, 16);
      l[r] = l[r] * corr + rs;
      m[r] = nm;
      o0[r] *= corr;
      o1[r] *= corr;
      const int row = r + hi16 * 8;
      plds[row * 32 + r15]      = (h16)p0;
      plds[row * 32 + 16 + r15] = (h16)p1;
    }
    asm volatile("s_wait_dscnt 0" ::: "memory");
    // P (16x32 f16) re-read in A layout from per-wave LDS
    const v16h pa = load_a(plds + r15 * 32, lane);
    // O += P x V : two channel halves; vbuf channel-major => contiguous keys per lane
    o0 = WMMA_F16(pa, load_b(vl + (size_t)r15 * 32 + hi16 * 16), o0);
    o1 = WMMA_F16(pa, load_b(vl + (size_t)(16 + r15) * 32 + hi16 * 16), o1);

    __syncthreads();         // all waves done reading before buffer reuse
  }

  const int b = bh >> 2, h = bh & 3;
  #pragma unroll
  for (int r = 0; r < 8; ++r) {
    const float inv = 1.0f / l[r];
    const int q = qt * 16 + r + hi16 * 8;
    h16* op = Ob + ((size_t)b * N_ + q) * HID_ + h * DH_;
    op[r15]      = (h16)(o0[r] * inv);
    op[16 + r15] = (h16)(o1[r] * inv);
  }
}

// ---------------- output projection: [256,128] x [128,4096] + bias ----------------
__global__ __launch_bounds__(128) void outproj_kernel(const h16* __restrict__ wo16,
                                                      const h16* __restrict__ Ob,
                                                      const float* __restrict__ bias,
                                                      float* __restrict__ y) {
  const int lane = threadIdx.x & 31;
  const int wid  = threadIdx.x >> 5;
  const int tile = blockIdx.x * 4 + wid;           // 4 * 16 * 256 = 16384
  const int b   = tile / (16 * 256);
  const int rem = tile - b * (16 * 256);
  const int ot  = rem >> 8;                        // 0..15
  const int nt  = rem & 255;
  const int r15 = lane & 15;
  const int hi16 = lane >> 4;

  const h16* X = Ob + (size_t)b * N_ * HID_;
  v8f acc = {};
  #pragma unroll
  for (int ks = 0; ks < 4; ++ks) {
    const int kb = ks * 32;
    const h16* arow = wo16 + (size_t)(ot * 16 + r15) * HID_ + kb;
    v16h a = load_a(arow, lane);
    const h16* bp = X + (size_t)(nt * 16 + r15) * HID_ + kb + hi16 * 16;
    v16h bm = load_b(bp);
    acc = WMMA_F16(a, bm, acc);
  }
  #pragma unroll
  for (int r = 0; r < 8; ++r) {
    const int o = ot * 16 + r + hi16 * 8;
    const int n = nt * 16 + r15;
    y[((size_t)b * C_ + o) * N_ + n] = acc[r] + bias[o];
  }
}

// ---------------- host launcher ----------------
extern "C" void kernel_launch(void* const* d_in, const int* in_sizes, int n_in,
                              void* d_out, int out_size, void* d_ws, size_t ws_size,
                              hipStream_t stream) {
  const float* x     = (const float*)d_in[0];
  const float* gamma = (const float*)d_in[1];
  const float* beta  = (const float*)d_in[2];
  const float* w_qkv = (const float*)d_in[3];
  const float* w_out = (const float*)d_in[4];
  const float* b_out = (const float*)d_in[5];
  float* y = (float*)d_out;

  char* w = (char*)d_ws;
  size_t off = 0;
  float* stats = (float*)(w + off); off += 256;
  h16* wq16 = (h16*)(w + off); off += (size_t)QKV_R * C_ * sizeof(h16);
  h16* wo16 = (h16*)(w + off); off += (size_t)C_ * HID_ * sizeof(h16);
  h16* xnT  = (h16*)(w + off); off += (size_t)B_ * N_ * C_ * sizeof(h16);
  h16* Qb   = (h16*)(w + off); off += (size_t)B_ * HEADS_ * N_ * DH_ * sizeof(h16);
  h16* Kb   = (h16*)(w + off); off += (size_t)B_ * HEADS_ * N_ * DH_ * sizeof(h16);
  h16* Vb   = (h16*)(w + off); off += (size_t)B_ * HEADS_ * DH_ * N_ * sizeof(h16);
  h16* Ob   = (h16*)(w + off); off += (size_t)B_ * N_ * HID_ * sizeof(h16);
  (void)ws_size; (void)in_sizes; (void)n_in; (void)out_size;

  prep_kernel<<<128, 256, 0, stream>>>(w_qkv, w_out, wq16, wo16, stats);
  stats_kernel<<<dim3(64, B_), 256, 0, stream>>>(x, stats);
  normalize_kernel<<<(B_ * N_) / 256, 256, 0, stream>>>(x, gamma, beta, stats, xnT);
  qkv_gemm_kernel<<<(B_ * 24 * 256) / 4, 128, 0, stream>>>(wq16, xnT, Qb, Kb, Vb);
  flash_attn_kernel<<<(B_ * HEADS_ * (N_ / 16)) / 4, 128, 0, stream>>>(Qb, Kb, Vb, Ob);
  outproj_kernel<<<(B_ * 16 * 256) / 4, 128, 0, stream>>>(wo16, Ob, b_out, y);
}
